// ChessGATv2_66623532695601
// MI455X (gfx1250) — compile-verified, matrix operand order checked by reading, not run on hardware
//
#include <hip/hip_runtime.h>
#include <math.h>

typedef __attribute__((ext_vector_type(16))) _Float16 v16h;
typedef __attribute__((ext_vector_type(8)))  _Float16 v8h;
typedef __attribute__((ext_vector_type(8)))  float    v8f;

#define NN   65536
#define EE   524288
#define GG   1024
#define NDIM 21
#define EDIM 12
#define HID  128
#define NH   4
#define CH   32
#define NL   8
#define EPS_BN 1e-5f
#define EP   (EE + NN)          // edges + self loops = 589824
#define KPOL 288                // 2*HID + EDIM = 268, padded to 9*32

// ---- CDNA5 async global->LDS staging (ASYNCcnt path) ----
__device__ __forceinline__ void async_to_lds_b128(const _Float16* g, _Float16* lds_generic){
  unsigned lds_off = (unsigned)(size_t)lds_generic;   // low 32 bits of generic = LDS byte addr
  asm volatile("global_load_async_to_lds_b128 %0, %1, off"
               :: "v"(lds_off), "v"(g) : "memory");
}
__device__ __forceinline__ void wait_async_then_barrier(){
  asm volatile("s_wait_asynccnt 0x0" ::: "memory");
  __syncthreads();
}

// ---------------- helpers: ordered-uint float max trick ----------------
__device__ __forceinline__ unsigned mapf(float x){
  unsigned u = __float_as_uint(x);
  return (u & 0x80000000u) ? ~u : (u | 0x80000000u);
}
__device__ __forceinline__ float unmapf(unsigned u){
  if (u == 0u) return 0.0f;   // node with no incoming edges: reference sets 0
  unsigned v = (u & 0x80000000u) ? (u & 0x7FFFFFFFu) : ~u;
  return __uint_as_float(v);
}

// ---------------- input projection: h = relu(x @ Win + bin) ----------------
__global__ void k_input_proj(const float* __restrict__ x, const float* __restrict__ Win,
                             const float* __restrict__ bin, float* __restrict__ h,
                             _Float16* __restrict__ h16){
  int tid = blockIdx.x * blockDim.x + threadIdx.x;
  if (tid >= NN * HID) return;
  int c = tid & (HID - 1), n = tid >> 7;
  float s = bin[c];
  const float* xr = x + n * NDIM;
  #pragma unroll
  for (int d = 0; d < NDIM; ++d) s += xr[d] * Win[d * HID + c];
  s = fmaxf(s, 0.f);
  h[tid] = s;
  h16[tid] = (_Float16)s;
}

// ---------------- degree + edge-attr mean (self-loop fill) ----------------
__global__ void k_deg_sum(const float* __restrict__ ea, const int* __restrict__ dst0,
                          float* __restrict__ deg, float* __restrict__ sum_ea){
  int e = blockIdx.x * blockDim.x + threadIdx.x;
  if (e >= EE) return;
  int d = dst0[e];
  atomicAdd(&deg[d], 1.f);
  #pragma unroll
  for (int k = 0; k < EDIM; ++k) atomicAdd(&sum_ea[d * EDIM + k], ea[e * EDIM + k]);
}
__global__ void k_mean_fin(float* __restrict__ mean_ea, const float* __restrict__ deg){
  int t = blockIdx.x * blockDim.x + threadIdx.x;
  if (t >= NN * EDIM) return;
  mean_ea[t] = mean_ea[t] / fmaxf(deg[t / EDIM], 1.f);
}

// ---------------- small conversions ----------------
__global__ void k_ea16(const float* __restrict__ ea, _Float16* __restrict__ ea16){
  int t = blockIdx.x * blockDim.x + threadIdx.x;
  if (t < EE * EDIM) ea16[t] = (_Float16)ea[t];
}
__global__ void k_cvt_w(const float* __restrict__ Wl, const float* __restrict__ Wr,
                        _Float16* __restrict__ wlt, _Float16* __restrict__ wrt){
  int t = blockIdx.x * blockDim.x + threadIdx.x;   // 128*128, transpose to [nc][k]
  if (t >= HID * HID) return;
  int nc = t >> 7, k = t & 127;
  wlt[nc * HID + k] = (_Float16)Wl[k * HID + nc];
  wrt[nc * HID + k] = (_Float16)Wr[k * HID + nc];
}
__global__ void k_cvt_wp1(const float* __restrict__ Wp1, _Float16* __restrict__ wt){
  int t = blockIdx.x * blockDim.x + threadIdx.x;   // 64 * 288, transposed + zero-pad
  if (t >= 64 * KPOL) return;
  int j = t / KPOL, k = t % KPOL;
  wt[t] = (k < 2 * HID + EDIM) ? (_Float16)Wp1[k * 64 + j] : (_Float16)0.f;
}

// ---- WMMA node GEMM: one wave computes a 16x128 strip; B staged in LDS ----
// grid.x = NN/(16*8) = 512 blocks, 256 threads (8 waves), grid.z selects Wl/Wr
__global__ void k_gemm_node(const _Float16* __restrict__ A, const _Float16* __restrict__ Bl,
                            const _Float16* __restrict__ Br, float* __restrict__ Cl,
                            float* __restrict__ Cr){
  __shared__ _Float16 Bs[HID * HID];               // 32 KB weight tile
  const _Float16* B = (blockIdx.z == 0) ? Bl : Br; // B stored transposed: [Nc][K]
  float* C = (blockIdx.z == 0) ? Cl : Cr;

  // cooperative async stage: 2048 x 16B chunks over 256 threads
  #pragma unroll
  for (int i = 0; i < 8; ++i){
    int idx16 = threadIdx.x + i * 256;
    async_to_lds_b128(B + idx16 * 8, &Bs[idx16 * 8]);
  }
  wait_async_then_barrier();

  int wave = threadIdx.x >> 5;
  int lane = threadIdx.x & 31;
  int hh = lane >> 4, ln = lane & 15;
  int mbase = (blockIdx.x * 8 + wave) * 16;
  int M = mbase + ln;

  v8f acc[8] = {};
  #pragma unroll
  for (int ks = 0; ks < 4; ++ks){
    int kb = ks * 32;
    union { v16h v; v8h h[2]; } a;
    // ISA A-layout: lane half hh covers K = kb+hh*8..+8 and kb+16+hh*8..+8
    a.h[0] = *(const v8h*)(A + (size_t)M * HID + kb + hh * 8);
    a.h[1] = *(const v8h*)(A + (size_t)M * HID + kb + 16 + hh * 8);
    #pragma unroll
    for (int nt = 0; nt < 8; ++nt){
      union { v16h v; v8h h[2]; } b;
      const _Float16* brow = &Bs[(nt * 16 + ln) * HID];
      b.h[0] = *(const v8h*)(brow + kb + hh * 8);
      b.h[1] = *(const v8h*)(brow + kb + 16 + hh * 8);
      acc[nt] = __builtin_amdgcn_wmma_f32_16x16x32_f16(false, a.v, false, b.v,
                                                       (short)0, acc[nt], false, false);
    }
  }
  #pragma unroll
  for (int nt = 0; nt < 8; ++nt)
    #pragma unroll
    for (int r = 0; r < 8; ++r)
      C[(size_t)(mbase + r + 8 * hh) * HID + nt * 16 + ln] = acc[nt][r];
}

// ---------------- GATv2 edge scores (pass 1): e = sum_c lrelu(m)*att, segment max ----------------
__global__ void k_edge_score(const float* __restrict__ xl, const float* __restrict__ xr,
                             const float* __restrict__ eattr, const float* __restrict__ mean_ea,
                             const int* __restrict__ ei, const float* __restrict__ We,
                             const float* __restrict__ att,
                             float* __restrict__ escore, unsigned* __restrict__ emax){
  int e = (blockIdx.x * blockDim.x + threadIdx.x) >> 5;   // one wave per edge
  if (e >= EP) return;
  int lane = threadIdx.x & 31;
  int src = (e < EE) ? ei[e] : (e - EE);
  int dst = (e < EE) ? ei[EE + e] : (e - EE);
  const float* ea = (e < EE) ? (eattr + (size_t)e * EDIM) : (mean_ea + (size_t)(e - EE) * EDIM);
  float ear[EDIM];
  #pragma unroll
  for (int d = 0; d < EDIM; ++d) ear[d] = ea[d];
  int hd = lane >> 3, sub = lane & 7;
  float s = 0.f;
  #pragma unroll
  for (int t = 0; t < 4; ++t){
    int idx = hd * CH + sub + 8 * t;
    float el = 0.f;
    #pragma unroll
    for (int d = 0; d < EDIM; ++d) el += ear[d] * We[d * HID + idx];
    float m = xl[(size_t)src * HID + idx] + xr[(size_t)dst * HID + idx] + el;
    m = (m > 0.f) ? m : 0.2f * m;     // leaky_relu(0.2)
    s += m * att[idx];
  }
  s += __shfl_xor(s, 1);
  s += __shfl_xor(s, 2);
  s += __shfl_xor(s, 4);
  if (sub == 0){
    escore[(size_t)e * NH + hd] = s;
    atomicMax(&emax[dst * NH + hd], mapf(s));
  }
}

// ---------------- softmax pass 2: p = exp(e - max), denom += p ----------------
__global__ void k_edge_soft(const int* __restrict__ ei, float* __restrict__ escore,
                            const unsigned* __restrict__ emax, float* __restrict__ denom){
  int t = blockIdx.x * blockDim.x + threadIdx.x;
  if (t >= EP * NH) return;
  int e = t >> 2, hd = t & 3;
  int dst = (e < EE) ? ei[EE + e] : (e - EE);
  float p = __expf(escore[t] - unmapf(emax[dst * NH + hd]));
  escore[t] = p;
  atomicAdd(&denom[dst * NH + hd], p);
}

// ---------------- aggregation: agg[dst] += alpha * xl[src] ----------------
__global__ void k_aggregate(const float* __restrict__ xl, const float* __restrict__ p,
                            const float* __restrict__ denom, const int* __restrict__ ei,
                            float* __restrict__ agg){
  int e = (blockIdx.x * blockDim.x + threadIdx.x) >> 5;   // one wave per edge
  if (e >= EP) return;
  int lane = threadIdx.x & 31;
  int src = (e < EE) ? ei[e] : (e - EE);
  int dst = (e < EE) ? ei[EE + e] : (e - EE);
  #pragma unroll
  for (int t = 0; t < 4; ++t){   // t == head (channels are head-contiguous)
    float alpha = p[(size_t)e * NH + t] / fmaxf(denom[dst * NH + t], 1e-16f);
    int idx = t * CH + lane;
    atomicAdd(&agg[(size_t)dst * HID + idx], alpha * xl[(size_t)src * HID + idx]);
  }
}

// ---------------- batchnorm: stats then apply (+bias, residual, relu) ----------------
__global__ void k_bn_stats(const float* __restrict__ agg, const float* __restrict__ bconv,
                           float* __restrict__ bnsum, float* __restrict__ bnsq){
  int c = threadIdx.x;                 // blockDim = 128
  int base = blockIdx.x * 256;
  float s = 0.f, q = 0.f, bc = bconv[c];
  for (int r = 0; r < 256; ++r){
    float v = agg[(size_t)(base + r) * HID + c] + bc;
    s += v; q += v * v;
  }
  atomicAdd(&bnsum[c], s);
  atomicAdd(&bnsq[c], q);
}
__global__ void k_bn_apply(const float* __restrict__ agg, const float* __restrict__ bconv,
                           const float* __restrict__ bnsum, const float* __restrict__ bnsq,
                           const float* __restrict__ scale, const float* __restrict__ bias,
                           const float* __restrict__ res, int odd,
                           float* __restrict__ h, _Float16* __restrict__ h16){
  int tid = blockIdx.x * blockDim.x + threadIdx.x;
  if (tid >= NN * HID) return;
  int c = tid & (HID - 1);
  float mu  = bnsum[c] * (1.f / NN);
  float var = bnsq[c] * (1.f / NN) - mu * mu;
  float v = (agg[tid] + bconv[c] - mu) * rsqrtf(var + EPS_BN) * scale[c] + bias[c];
  if (odd) v += res[tid];
  v = fmaxf(v, 0.f);
  h[tid] = v;
  h16[tid] = (_Float16)v;
}

// ---------------- mean pooling + value head ----------------
__global__ void k_pool(const float* __restrict__ h, const int* __restrict__ batch,
                       float* __restrict__ pooled, float* __restrict__ counts){
  int tid = blockIdx.x * blockDim.x + threadIdx.x;
  if (tid >= NN * HID) return;
  int c = tid & (HID - 1), n = tid >> 7;
  int g = batch[n];
  atomicAdd(&pooled[(size_t)g * HID + c], h[tid]);
  if (c == 0) atomicAdd(&counts[g], 1.f);
}
__global__ void k_value(const float* __restrict__ pooled, const float* __restrict__ counts,
                        const float* __restrict__ Wv1, const float* __restrict__ bv1,
                        const float* __restrict__ Wv2, const float* __restrict__ bv2,
                        float* __restrict__ out){
  __shared__ float hid[64];
  int g = blockIdx.x, j = threadIdx.x;   // 64 threads
  float inv = 1.f / fmaxf(counts[g], 1.f);
  float s = bv1[j];
  for (int d = 0; d < HID; ++d) s += pooled[(size_t)g * HID + d] * inv * Wv1[d * 64 + j];
  hid[j] = fmaxf(s, 0.f);
  __syncthreads();
  if (j < 3){
    float o = bv2[j];
    #pragma unroll 8
    for (int k = 0; k < 64; ++k) o += hid[k] * Wv2[k * 3 + j];
    out[g * 3 + j] = o;
  }
}

// ---------------- WMMA policy head: [h[src]|h[dst]|ea] @ Wp1 -> relu -> @ Wp2 ----------------
__device__ __forceinline__ v8h load8_edge(const _Float16* __restrict__ h16,
                                          const _Float16* __restrict__ ea16,
                                          int src, int dst, int e, int K){
  if (K < HID)     return *(const v8h*)(h16 + (size_t)src * HID + K);
  if (K < 2 * HID) return *(const v8h*)(h16 + (size_t)dst * HID + (K - HID));
  v8h r;
  #pragma unroll
  for (int t = 0; t < 8; ++t){
    int kk = K - 2 * HID + t;
    r[t] = (kk < EDIM) ? ea16[(size_t)e * EDIM + kk] : (_Float16)0.f;
  }
  return r;
}

__global__ void k_policy(const _Float16* __restrict__ h16, const _Float16* __restrict__ ea16,
                         const int* __restrict__ ei, const _Float16* __restrict__ wp1t,
                         const float* __restrict__ bp1, const float* __restrict__ wp2,
                         const float* __restrict__ bp2, float* __restrict__ out){
  __shared__ _Float16 Ws[64 * KPOL];   // 36 KB staged weights
  #pragma unroll
  for (int i = 0; i < 9; ++i){         // 2304 x 16B chunks over 256 threads
    int idx16 = threadIdx.x + i * 256;
    async_to_lds_b128(wp1t + idx16 * 8, &Ws[idx16 * 8]);
  }
  wait_async_then_barrier();

  int wave = (blockIdx.x * blockDim.x + threadIdx.x) >> 5;   // 16 edges per wave
  int lane = threadIdx.x & 31;
  int hh = lane >> 4, ln = lane & 15;
  int e0 = wave * 16;
  int erow = e0 + ln;
  int src = ei[erow], dst = ei[EE + erow];
  v8f acc[4] = {};
  for (int ks = 0; ks < 9; ++ks){
    int kb = ks * 32;
    union { v16h v; v8h h[2]; } a;
    a.h[0] = load8_edge(h16, ea16, src, dst, erow, kb + hh * 8);
    a.h[1] = load8_edge(h16, ea16, src, dst, erow, kb + 16 + hh * 8);
    #pragma unroll
    for (int t = 0; t < 4; ++t){
      union { v16h v; v8h h[2]; } b;
      const _Float16* brow = &Ws[(t * 16 + ln) * KPOL];
      b.h[0] = *(const v8h*)(brow + kb + hh * 8);
      b.h[1] = *(const v8h*)(brow + kb + 16 + hh * 8);
      acc[t] = __builtin_amdgcn_wmma_f32_16x16x32_f16(false, a.v, false, b.v,
                                                      (short)0, acc[t], false, false);
    }
  }
  float b2 = bp2[0];
  #pragma unroll
  for (int r = 0; r < 8; ++r){
    float part = 0.f;
    #pragma unroll
    for (int t = 0; t < 4; ++t){
      int jcol = t * 16 + ln;
      part += fmaxf(acc[t][r] + bp1[jcol], 0.f) * wp2[jcol];
    }
    part += __shfl_xor(part, 1);
    part += __shfl_xor(part, 2);
    part += __shfl_xor(part, 4);
    part += __shfl_xor(part, 8);
    if (ln == 0) out[e0 + r + 8 * hh] = part + b2;   // lanes 0 / 16 cover M=r / M=r+8
  }
}

// ---------------- host orchestration ----------------
extern "C" void kernel_launch(void* const* d_in, const int* in_sizes, int n_in,
                              void* d_out, int out_size, void* d_ws, size_t ws_size,
                              hipStream_t stream){
  const float* x     = (const float*)d_in[0];
  const float* ea    = (const float*)d_in[1];
  const int*   ei    = (const int*)d_in[2];     // [2,E]
  const int*   batch = (const int*)d_in[3];
  const float* Win   = (const float*)d_in[4];
  const float* bin   = (const float*)d_in[5];
  const float* Wl    = (const float*)d_in[6];
  const float* Wr    = (const float*)d_in[7];
  const float* We    = (const float*)d_in[8];
  const float* att   = (const float*)d_in[9];
  const float* bconv = (const float*)d_in[10];
  const float* bns   = (const float*)d_in[11];
  const float* bnb   = (const float*)d_in[12];
  const float* Wv1   = (const float*)d_in[13];
  const float* bv1   = (const float*)d_in[14];
  const float* Wv2   = (const float*)d_in[15];
  const float* bv2   = (const float*)d_in[16];
  const float* Wp1   = (const float*)d_in[17];
  const float* bp1   = (const float*)d_in[18];
  const float* Wp2   = (const float*)d_in[19];
  const float* bp2   = (const float*)d_in[20];
  float* outv = (float*)d_out;          // value_logits [G,3]
  float* outp = outv + GG * 3;          // policy_logits [E]

  char* w = (char*)d_ws;
  auto alloc = [&](size_t bytes)->char*{ char* p = w; w += (bytes + 255) & ~size_t(255); return p; };
  float*    h      = (float*)   alloc((size_t)NN * HID * 4);
  float*    res    = (float*)   alloc((size_t)NN * HID * 4);
  float*    xl     = (float*)   alloc((size_t)NN * HID * 4);
  float*    xr     = (float*)   alloc((size_t)NN * HID * 4);
  float*    agg    = (float*)   alloc((size_t)NN * HID * 4);
  _Float16* h16    = (_Float16*)alloc((size_t)NN * HID * 2);
  float*    escore = (float*)   alloc((size_t)EP * NH * 4);
  unsigned* emax   = (unsigned*)alloc((size_t)NN * NH * 4);
  float*    denom  = (float*)   alloc((size_t)NN * NH * 4);
  float*    deg    = (float*)   alloc((size_t)NN * 4);
  float*    meanea = (float*)   alloc((size_t)NN * EDIM * 4);
  _Float16* ea16   = (_Float16*)alloc((size_t)EE * EDIM * 2);
  _Float16* wlt    = (_Float16*)alloc((size_t)HID * HID * 2);
  _Float16* wrt    = (_Float16*)alloc((size_t)HID * HID * 2);
  _Float16* wp1t   = (_Float16*)alloc((size_t)64 * KPOL * 2);
  float*    bnsum  = (float*)   alloc(HID * 4);
  float*    bnsq   = (float*)   alloc(HID * 4);
  float*    pooled = (float*)   alloc((size_t)GG * HID * 4);
  float*    counts = (float*)   alloc((size_t)GG * 4);

  // self-loop edge-attr mean + conversions
  hipMemsetAsync(deg, 0, (size_t)NN * 4, stream);
  hipMemsetAsync(meanea, 0, (size_t)NN * EDIM * 4, stream);
  k_deg_sum<<<(EE + 255) / 256, 256, 0, stream>>>(ea, ei + EE, deg, meanea);
  k_mean_fin<<<(NN * EDIM + 255) / 256, 256, 0, stream>>>(meanea, deg);
  k_ea16<<<(EE * EDIM + 255) / 256, 256, 0, stream>>>(ea, ea16);
  k_cvt_wp1<<<(64 * KPOL + 255) / 256, 256, 0, stream>>>(Wp1, wp1t);
  k_input_proj<<<NN * HID / 256, 256, 0, stream>>>(x, Win, bin, h, h16);

  for (int l = 0; l < NL; ++l){
    int odd = l & 1;
    if (!odd) hipMemcpyAsync(res, h, (size_t)NN * HID * 4, hipMemcpyDeviceToDevice, stream);
    k_cvt_w<<<(HID * HID + 255) / 256, 256, 0, stream>>>(Wl + (size_t)l * HID * HID,
                                                         Wr + (size_t)l * HID * HID, wlt, wrt);
    dim3 gg(512, 1, 2);   // z selects Wl->xl vs Wr->xr; each wave does a 16x128 strip
    k_gemm_node<<<gg, 256, 0, stream>>>(h16, wlt, wrt, xl, xr);
    hipMemsetAsync(emax, 0, (size_t)NN * NH * 4, stream);
    hipMemsetAsync(denom, 0, (size_t)NN * NH * 4, stream);
    hipMemsetAsync(agg, 0, (size_t)NN * HID * 4, stream);
    k_edge_score<<<EP / 8, 256, 0, stream>>>(xl, xr, ea, meanea, ei,
        We + (size_t)l * EDIM * HID, att + (size_t)l * HID, escore, emax);
    k_edge_soft<<<(EP * NH) / 256, 256, 0, stream>>>(ei, escore, emax, denom);
    k_aggregate<<<EP / 8, 256, 0, stream>>>(xl, escore, denom, ei, agg);
    hipMemsetAsync(bnsum, 0, HID * 4, stream);
    hipMemsetAsync(bnsq, 0, HID * 4, stream);
    k_bn_stats<<<NN / 256, 128, 0, stream>>>(agg, bconv + (size_t)l * HID, bnsum, bnsq);
    k_bn_apply<<<NN * HID / 256, 256, 0, stream>>>(agg, bconv + (size_t)l * HID, bnsum, bnsq,
        bns + (size_t)l * HID, bnb + (size_t)l * HID, res, odd, h, h16);
  }

  hipMemsetAsync(pooled, 0, (size_t)GG * HID * 4, stream);
  hipMemsetAsync(counts, 0, (size_t)GG * 4, stream);
  k_pool<<<NN * HID / 256, 256, 0, stream>>>(h, batch, pooled, counts);
  k_value<<<GG, 64, 0, stream>>>(pooled, counts, Wv1, bv1, Wv2, bv2, outv);
  k_policy<<<(EE / 16) / 8, 256, 0, stream>>>(h16, ea16, ei, wp1t, bp1, Wp2, bp2, outp);
}
